// MoELayer_31507880083722
// MI455X (gfx1250) — matrix operand order repaired and checked by reading, not compiled, not run on hardware
//
#include <hip/hip_runtime.h>
#include <hip/hip_bf16.h>
#include <math.h>

// ---------------- problem constants ----------------
#define D_MODEL 1024
#define NEXP    8
#define HID     8192       // w1 output width
#define FF      4096       // half of HID (SwiGLU)
#define SEQ     2048
#define MT      64         // M tile (rows per block tile)
#define KC      256        // K chunk staged in LDS
#define MAXROWS 4608       // 4096 pairs + 8*63 padding, rounded to 64

typedef __bf16 bf16;
typedef __attribute__((ext_vector_type(16))) __bf16 v16bf;
typedef __attribute__((ext_vector_type(8)))  float  v8f;
typedef __attribute__((ext_vector_type(4)))  int    v4i;

// Async global->LDS path (gfx1250): probe via __has_builtin, fall back to
// synchronous copies if this toolchain doesn't expose the builtins.
#if defined(__gfx1250__) && __has_builtin(__builtin_amdgcn_global_load_async_to_lds_b128) && __has_builtin(__builtin_amdgcn_s_wait_asynccnt)
#define USE_ASYNC_LDS 1
#else
#define USE_ASYNC_LDS 0
#endif

// ---------------- workspace layout (bytes) ----------------
// 0      : counts  int[8]
// 64     : fill    int[8]
// 128    : base    int[16]   (prefix offsets, base[8] = total rows)
// 256    : tokIdx  int[2*SEQ]
// 16640  : tokW    float[2*SEQ]
// 33024  : rowTok  int[MAXROWS]
// 51456  : rowW    float[MAXROWS]
// 131072 : act     bf16[MAXROWS * FF]   (~38 MB)

// ============ 1) init lists ============
__global__ void init_lists_kernel(int* counts, int* fill, int* rowTok) {
    int t = blockIdx.x * blockDim.x + threadIdx.x;
    if (t < NEXP) { counts[t] = 0; fill[t] = 0; }
    if (t < MAXROWS) rowTok[t] = -1;
}

// ============ 2) gating: wave per token ============
__global__ __launch_bounds__(256) void gate_kernel(
    const float* __restrict__ x, const float* __restrict__ gw,
    const float* __restrict__ gb, int* __restrict__ tokIdx,
    float* __restrict__ tokW, int* __restrict__ counts)
{
    int tok  = (blockIdx.x * blockDim.x + threadIdx.x) >> 5;
    int lane = threadIdx.x & 31;
    if (tok >= SEQ) return;

    const float* xr = x + (size_t)tok * D_MODEL;
    float acc[NEXP];
#pragma unroll
    for (int e = 0; e < NEXP; ++e) acc[e] = 0.f;

    for (int d = lane; d < D_MODEL; d += 32) {
        float xv = xr[d];
        const float* g = gw + (size_t)d * NEXP;
#pragma unroll
        for (int e = 0; e < NEXP; ++e) acc[e] += xv * g[e];
    }
#pragma unroll
    for (int off = 16; off > 0; off >>= 1)
#pragma unroll
        for (int e = 0; e < NEXP; ++e) acc[e] += __shfl_xor(acc[e], off, 32);

    if (lane == 0) {
        float v[NEXP];
#pragma unroll
        for (int e = 0; e < NEXP; ++e) v[e] = acc[e] + gb[e];
        int i0 = 0;
#pragma unroll
        for (int e = 1; e < NEXP; ++e) if (v[e] > v[i0]) i0 = e;
        int i1 = (i0 == 0) ? 1 : 0;
#pragma unroll
        for (int e = 0; e < NEXP; ++e) if (e != i0 && v[e] > v[i1]) i1 = e;
        float e1 = __expf(v[i1] - v[i0]);       // v[i0] is the max
        float w0 = 1.f / (1.f + e1);
        float w1 = e1 * w0;
        tokIdx[2 * tok]     = i0;
        tokIdx[2 * tok + 1] = i1;
        tokW[2 * tok]       = w0;
        tokW[2 * tok + 1]   = w1;
        atomicAdd(&counts[i0], 1);
        atomicAdd(&counts[i1], 1);
    }
}

// ============ 3) prefix offsets (padded to MT) ============
__global__ void offsets_kernel(const int* counts, int* base) {
    int b = 0;
#pragma unroll
    for (int e = 0; e < NEXP; ++e) {
        base[e] = b;
        b += (counts[e] + MT - 1) & ~(MT - 1);
    }
    base[NEXP] = b;
}

// ============ 4) scatter tokens into per-expert rows ============
__global__ void scatter_kernel(const int* __restrict__ tokIdx,
                               const float* __restrict__ tokW,
                               const int* __restrict__ base, int* fill,
                               int* __restrict__ rowTok, float* __restrict__ rowW)
{
    int t = blockIdx.x * blockDim.x + threadIdx.x;
    if (t >= SEQ) return;
#pragma unroll
    for (int k = 0; k < 2; ++k) {
        int e = tokIdx[2 * t + k];
        int slot = atomicAdd(&fill[e], 1);
        int row = base[e] + slot;
        rowTok[row] = t;
        rowW[row]   = tokW[2 * t + k];
    }
}

// ============ 5) out = sum_k w_k * b2[idx_k]  (weights sum to 1) ============
__global__ void init_out_kernel(const int* __restrict__ tokIdx,
                                const float* __restrict__ tokW,
                                const float* __restrict__ b2,
                                float* __restrict__ out)
{
    int i = blockIdx.x * blockDim.x + threadIdx.x;
    if (i >= SEQ * D_MODEL) return;
    int t = i >> 10, d = i & (D_MODEL - 1);
    out[i] = tokW[2 * t]     * b2[(size_t)tokIdx[2 * t]     * D_MODEL + d]
           + tokW[2 * t + 1] * b2[(size_t)tokIdx[2 * t + 1] * D_MODEL + d];
}

// ============ 6) GEMM1: h = x @ w1 + b1 ; act = silu(h_g) * h_x ============
// block = 256 thr = 8 waves; each wave OWNS a 16-col group (no duplicated B
// loads) and computes M=64 rows via 4 A-fragments -> 8 WMMAs per k-step.
// Block tile: 64 rows x 128 pair-cols (x-half and gate-half both computed).
__global__ __launch_bounds__(256) void gemm1_kernel(
    const float* __restrict__ x, const float* __restrict__ w1,
    const float* __restrict__ b1, const int* __restrict__ base,
    const int* __restrict__ rowTok, bf16* __restrict__ act)
{
    __shared__ bf16 As[MT][KC + 8];         // 64 x 264 x 2B = 33 KB

    int row0 = blockIdx.y * MT;
    if (row0 >= base[NEXP]) return;
    int e = 0;
#pragma unroll
    for (int i = 1; i < NEXP; ++i) if (row0 >= base[i]) e = i;

    int tid  = threadIdx.x;
    int wave = tid >> 5, lane = tid & 31;
    int n0  = blockIdx.x * 128 + wave * 16; // column group in [0, FF)
    int col = n0 + (lane & 15);
    int kh  = lane >> 4;                    // half of wave (K sub-select)

    const float* w1e = w1 + (size_t)e * D_MODEL * HID;
    const float* b1e = b1 + (size_t)e * HID;

    v8f cx[4] = {}, cg[4] = {};

    // cooperative staging: 4 threads per row, 64 cols each (f32 -> bf16)
    int srow = tid >> 2;
    int scol = (tid & 3) * 64;
    int gtok = rowTok[row0 + srow];

    for (int kc = 0; kc < D_MODEL; kc += KC) {
        __syncthreads();
        if (gtok >= 0) {
            const float* xr = x + (size_t)gtok * D_MODEL + kc + scol;
#pragma unroll
            for (int j = 0; j < 64; j += 4) {
                float4 v = *(const float4*)(xr + j);
                bf16* d = &As[srow][scol + j];
                d[0] = (bf16)v.x; d[1] = (bf16)v.y; d[2] = (bf16)v.z; d[3] = (bf16)v.w;
            }
        } else {
#pragma unroll
            for (int j = 0; j < 64; j += 4) {
                bf16* d = &As[srow][scol + j];
                d[0] = (bf16)0.f; d[1] = (bf16)0.f; d[2] = (bf16)0.f; d[3] = (bf16)0.f;
            }
        }
        __syncthreads();

#pragma unroll
        for (int ks = 0; ks < KC; ks += 32) {
            int kg = kc + ks + 16 * kh;     // B: lane col = n, K = 16*kh + i
            __builtin_prefetch(&w1e[(size_t)(kg + 32) * HID + col], 0, 3);
            v16bf bx, bg;
#pragma unroll
            for (int i = 0; i < 16; ++i) {
                bx[i] = (bf16)w1e[(size_t)(kg + i) * HID + col];
                bg[i] = (bf16)w1e[(size_t)(kg + i) * HID + (col + FF)];
            }
#pragma unroll
            for (int m = 0; m < 4; ++m) {
                // A fragment: lane row = lane&15, K = 16*kh..+15
                union { int4 q[2]; v16bf v; } a;
                const bf16* ap = &As[m * 16 + (lane & 15)][ks + 16 * kh];
                a.q[0] = *(const int4*)ap;
                a.q[1] = *(const int4*)(ap + 8);
                cx[m] = __builtin_amdgcn_wmma_f32_16x16x32_bf16(false, a.v, false, bx,
                                                                (short)0, cx[m], false, false);
                cg[m] = __builtin_amdgcn_wmma_f32_16x16x32_bf16(false, a.v, false, bg,
                                                                (short)0, cg[m], false, false);
            }
        }
    }

    float bx_bias = b1e[col];
    float bg_bias = b1e[col + FF];
#pragma unroll
    for (int m = 0; m < 4; ++m) {
#pragma unroll
        for (int v = 0; v < 8; ++v) {       // C/D layout: row = v + 8*kh, col = lane&15
            int r = row0 + m * 16 + v + 8 * kh;
            float hg = cg[m][v] + bg_bias;
            float hx = cx[m][v] + bx_bias;
            float a  = hx * (hg / (1.f + __expf(-hg)));   // silu(hg) * hx
            act[(size_t)r * FF + col] = (bf16)a;
        }
    }
}

// ============ 7) GEMM2: y = act @ w2 ; out[tok] += w * y ============
// Same structure: 8 waves own 16-col groups, M=64, 4 WMMAs per 16 B loads.
// act is already bf16 -> stage it with async global->LDS copies (ASYNCcnt).
__global__ __launch_bounds__(256) void gemm2_kernel(
    const bf16* __restrict__ act, const float* __restrict__ w2,
    const int* __restrict__ base, const int* __restrict__ rowTok,
    const float* __restrict__ rowW, float* __restrict__ out)
{
    __shared__ bf16 As[MT][KC + 8];

    int row0 = blockIdx.y * MT;
    if (row0 >= base[NEXP]) return;
    int e = 0;
#pragma unroll
    for (int i = 1; i < NEXP; ++i) if (row0 >= base[i]) e = i;

    int tid  = threadIdx.x;
    int wave = tid >> 5, lane = tid & 31;
    int n0  = blockIdx.x * 128 + wave * 16; // column group in [0, D_MODEL)
    int col = n0 + (lane & 15);
    int kh  = lane >> 4;

    const float* w2e = w2 + (size_t)e * FF * D_MODEL;
    v8f c[4] = {};

    int srow = tid >> 2;
    int scol = (tid & 3) * 64;
    const bf16* ar = act + (size_t)(row0 + srow) * FF + scol;

    for (int kc = 0; kc < FF; kc += KC) {
        __syncthreads();
#if USE_ASYNC_LDS
#pragma unroll
        for (int j = 0; j < 64; j += 8) {   // 8 bf16 = 16 B per async b128
            __builtin_amdgcn_global_load_async_to_lds_b128(
                (__attribute__((address_space(1))) v4i*)(ar + kc + j),
                (__attribute__((address_space(3))) v4i*)&As[srow][scol + j],
                0, 0);
        }
        __builtin_amdgcn_s_wait_asynccnt(0);
#else
#pragma unroll
        for (int j = 0; j < 64; j += 8) {   // bf16: 8 elems = 16 B
            int4 v = *(const int4*)(ar + kc + j);
            *(int4*)&As[srow][scol + j] = v;
        }
#endif
        __syncthreads();

#pragma unroll
        for (int ks = 0; ks < KC; ks += 32) {
            int kg = kc + ks + 16 * kh;
            __builtin_prefetch(&w2e[(size_t)(kg + 32) * D_MODEL + col], 0, 3);
            v16bf b;
#pragma unroll
            for (int i = 0; i < 16; ++i)
                b[i] = (bf16)w2e[(size_t)(kg + i) * D_MODEL + col];
#pragma unroll
            for (int m = 0; m < 4; ++m) {
                union { int4 q[2]; v16bf v; } a;
                const bf16* ap = &As[m * 16 + (lane & 15)][ks + 16 * kh];
                a.q[0] = *(const int4*)ap;
                a.q[1] = *(const int4*)(ap + 8);
                c[m] = __builtin_amdgcn_wmma_f32_16x16x32_bf16(false, a.v, false, b,
                                                               (short)0, c[m], false, false);
            }
        }
    }

#pragma unroll
    for (int m = 0; m < 4; ++m) {
#pragma unroll
        for (int v = 0; v < 8; ++v) {
            int r = row0 + m * 16 + v + 8 * kh;
            int tok = rowTok[r];
            if (tok >= 0)
                atomicAdd(&out[(size_t)tok * D_MODEL + col], rowW[r] * c[m][v]);
        }
    }
}

// ================= launcher =================
extern "C" void kernel_launch(void* const* d_in, const int* in_sizes, int n_in,
                              void* d_out, int out_size, void* d_ws, size_t ws_size,
                              hipStream_t stream) {
    const float* x  = (const float*)d_in[0];
    const float* gw = (const float*)d_in[1];
    const float* gb = (const float*)d_in[2];
    const float* w1 = (const float*)d_in[3];
    const float* b1 = (const float*)d_in[4];
    const float* w2 = (const float*)d_in[5];
    const float* b2 = (const float*)d_in[6];
    float* out = (float*)d_out;

    char* ws = (char*)d_ws;
    int*   counts = (int*)(ws + 0);
    int*   fill   = (int*)(ws + 64);
    int*   base   = (int*)(ws + 128);
    int*   tokIdx = (int*)(ws + 256);
    float* tokW   = (float*)(ws + 16640);
    int*   rowTok = (int*)(ws + 33024);
    float* rowW   = (float*)(ws + 51456);
    bf16*  act    = (bf16*)(ws + 131072);

    init_lists_kernel<<<(MAXROWS + 255) / 256, 256, 0, stream>>>(counts, fill, rowTok);
    gate_kernel<<<SEQ / 8, 256, 0, stream>>>(x, gw, gb, tokIdx, tokW, counts);
    offsets_kernel<<<1, 1, 0, stream>>>(counts, base);
    scatter_kernel<<<SEQ / 256, 256, 0, stream>>>(tokIdx, tokW, base, fill, rowTok, rowW);
    init_out_kernel<<<(SEQ * D_MODEL) / 256, 256, 0, stream>>>(tokIdx, tokW, b2, out);

    dim3 g1(FF / 128, MAXROWS / MT);          // 32 x 72
    gemm1_kernel<<<g1, 256, 0, stream>>>(x, w1, b1, base, rowTok, act);

    dim3 g2(D_MODEL / 128, MAXROWS / MT);     // 8 x 72
    gemm2_kernel<<<g2, 256, 0, stream>>>(act, w2, base, rowTok, rowW, out);
}
